// Model_15951508538245
// MI455X (gfx1250) — matrix operand order repaired and checked by reading, not compiled, not run on hardware
//
#include <hip/hip_runtime.h>
#include <hip/hip_bf16.h>

typedef _Float16 half_t;
typedef __attribute__((ext_vector_type(16))) _Float16 v16h;
typedef __attribute__((ext_vector_type(8)))  _Float16 v8h;
typedef __attribute__((ext_vector_type(8)))  float    v8f;

union V16 { v16h v; v8h h[2]; };

// ---------------------------------------------------------------------------
// Weight convert (f32 -> f16, with K padding for layer 1): dst is [Cout, Cin_pad]
// ---------------------------------------------------------------------------
__global__ void convert_w_kernel(const float* __restrict__ src, half_t* __restrict__ dst,
                                 int Cout, int Cin_real, int Cin_pad) {
    int idx = blockIdx.x * 256 + threadIdx.x;
    if (idx >= Cout * Cin_pad) return;
    int k = idx % Cin_pad, o = idx / Cin_pad;
    float v = (k < Cin_real) ? src[o * Cin_real + k] : 0.0f;
    dst[idx] = (half_t)v;
}

__global__ void zero_f32_kernel(float* __restrict__ p, int n) {
    int i = blockIdx.x * 256 + threadIdx.x;
    if (i < n) p[i] = 0.0f;
}

// ---------------------------------------------------------------------------
// Fused WMMA GEMM:  H[P,Cout] = transform_in(A[P,Cin_real]) * W16^T (+bias, relu)
// transform_in: optional per-channel v*scale+bias then relu (folded BN of prev)
// optional per-column sum / sum^2 stats (for this layer's BN)
//
// block = 128 threads = 4 waves; block tile = 64 rows x 64 cols.
// Wave tile = 32x32 (2 A-frags x 2 B-frags -> 4 accumulators): each fragment
// feeds two WMMAs. A and W panels both staged in LDS (KP K-columns at a time),
// so the inner loop is ds_load_b128 + v_wmma only; global weight traffic is
// one coalesced panel copy per block.
// ---------------------------------------------------------------------------
__global__ void wmma_gemm_kernel(const float* __restrict__ A,
                                 const float* __restrict__ a_scale,
                                 const float* __restrict__ a_bias,
                                 int a_relu, int Cin_real, int Cin_pad, int KP,
                                 const half_t* __restrict__ W16, int Cout, int P,
                                 const float* __restrict__ out_bias, int out_relu,
                                 float* __restrict__ H,
                                 float* __restrict__ col_sum,
                                 float* __restrict__ col_sumsq, int do_stats) {
    extern __shared__ char smem[];
    half_t* As = (half_t*)smem;                 // [64][KP] activations (f16)
    half_t* Ws = As + (size_t)64 * KP;          // [64][KP] weights (f16)
    float*  s_sum = (float*)(Ws + (size_t)64 * KP);  // [64]
    float*  s_sq  = s_sum + 64;                      // [64]

    const int tid  = threadIdx.x;
    const int m0   = blockIdx.x * 64;
    const int n0   = blockIdx.y * 64;
    const int lane = tid & 31;
    const int wave = tid >> 5;
    const int mw   = (wave >> 1) * 32;          // wave row offset in block tile
    const int nw   = (wave & 1) * 32;           // wave col offset in block tile
    const int base  = (lane < 16) ? 0 : 8;      // A-frag K sub-offset (ISA layout)
    const int kbase = (lane < 16) ? 0 : 8;      // B-frag K sub-offset (ISA layout)
    const int ln    = lane & 15;

    if (do_stats && tid < 64) { s_sum[tid] = 0.0f; s_sq[tid] = 0.0f; }

    v8f acc[4] = {};                            // [mi*2+ni]

    for (int kp = 0; kp < Cin_pad; kp += KP) {
        __syncthreads();
        // Stage A panel (64 rows x KP) with input transform, f32 -> f16.
        for (int idx = tid; idx < 64 * KP; idx += 128) {
            int r = idx / KP;
            int k = idx - r * KP;
            int gk = kp + k;
            float v = 0.0f;
            if (gk < Cin_real && (m0 + r) < P) {
                v = A[(size_t)(m0 + r) * Cin_real + gk];
                if (a_scale) v = v * a_scale[gk] + a_bias[gk];
                if (a_relu)  v = fmaxf(v, 0.0f);
            }
            As[idx] = (half_t)v;
        }
        // Stage W panel (64 rows x KP), vectorized 16B copies.
        const int KP8 = KP >> 3;
        for (int idx = tid; idx < 64 * KP8; idx += 128) {
            int r  = idx / KP8;
            int c8 = idx - r * KP8;
            *(v8h*)(Ws + (size_t)r * KP + c8 * 8) =
                *(const v8h*)(W16 + (size_t)(n0 + r) * Cin_pad + kp + c8 * 8);
        }
        __syncthreads();

        const half_t* rA0 = As + (size_t)(mw + ln) * KP;
        const half_t* rA1 = rA0 + (size_t)16 * KP;
        const half_t* rW0 = Ws + (size_t)(nw + ln) * KP;
        const half_t* rW1 = rW0 + (size_t)16 * KP;

        for (int kk = 0; kk < KP; kk += 32) {
            V16 a0, a1, b0, b1;
            a0.h[0] = *(const v8h*)(rA0 + kk + base);
            a0.h[1] = *(const v8h*)(rA0 + kk + 16 + base);
            a1.h[0] = *(const v8h*)(rA1 + kk + base);
            a1.h[1] = *(const v8h*)(rA1 + kk + 16 + base);
            b0.h[0] = *(const v8h*)(rW0 + kk + kbase);
            b0.h[1] = *(const v8h*)(rW0 + kk + kbase + 16);
            b1.h[0] = *(const v8h*)(rW1 + kk + kbase);
            b1.h[1] = *(const v8h*)(rW1 + kk + kbase + 16);
            acc[0] = __builtin_amdgcn_wmma_f32_16x16x32_f16(false, a0.v, false, b0.v, (short)0, acc[0], false, false);
            acc[1] = __builtin_amdgcn_wmma_f32_16x16x32_f16(false, a0.v, false, b1.v, (short)0, acc[1], false, false);
            acc[2] = __builtin_amdgcn_wmma_f32_16x16x32_f16(false, a1.v, false, b0.v, (short)0, acc[2], false, false);
            acc[3] = __builtin_amdgcn_wmma_f32_16x16x32_f16(false, a1.v, false, b1.v, (short)0, acc[3], false, false);
        }
    }

    // Output + per-column stats. D rows: r + 8*(lane>=16).
    const int mo = (lane >> 4) << 3;
    for (int t = 0; t < 4; ++t) {
        int mi = t >> 1, ni = t & 1;
        int n  = n0 + nw + ni * 16 + ln;
        float ob = out_bias ? out_bias[n] : 0.0f;
        int mbase = m0 + mw + mi * 16 + mo;
        float lsum = 0.0f, lsq = 0.0f;
        for (int r = 0; r < 8; ++r) {
            float v = acc[t][r] + ob;
            if (do_stats) { lsum += v; lsq += v * v; }
            if (out_relu) v = fmaxf(v, 0.0f);
            if (mbase + r < P) H[(size_t)(mbase + r) * Cout + n] = v;
        }
        if (do_stats) {
            int col = nw + ni * 16 + ln;
            atomicAdd(&s_sum[col], lsum);   // ds_add_f32
            atomicAdd(&s_sq[col],  lsq);
        }
    }
    if (do_stats) {
        __syncthreads();
        if (tid < 64) {
            atomicAdd(&col_sum[n0 + tid],   s_sum[tid]);
            atomicAdd(&col_sumsq[n0 + tid], s_sq[tid]);
        }
    }
}

// ---------------------------------------------------------------------------
// BN finalize: fold mean/var/gamma/beta into per-channel scale/bias.
// ---------------------------------------------------------------------------
__global__ void bn_finalize_kernel(const float* __restrict__ sum, const float* __restrict__ sq,
                                   const float* __restrict__ g, const float* __restrict__ b,
                                   float* __restrict__ scale, float* __restrict__ bias,
                                   int C, float invP) {
    int c = blockIdx.x * 256 + threadIdx.x;
    if (c >= C) return;
    float mu  = sum[c] * invP;
    float var = sq[c] * invP - mu * mu;
    float s   = g[c] * rsqrtf(var + 1e-5f);
    scale[c] = s;
    bias[c]  = b[c] - mu * s;
}

// ---------------------------------------------------------------------------
// Max-pool over N with BN+ReLU on the fly. grid (B, 16): each block reduces
// 256 points x 256 channels; cross-block combine via atomicMax on float bits
// (valid: post-ReLU values are >= 0, so IEEE bits are monotonic).
// ---------------------------------------------------------------------------
__global__ void maxpool_kernel(const float* __restrict__ H5,
                               const float* __restrict__ scale,
                               const float* __restrict__ bias,
                               float* __restrict__ z) {
    int b = blockIdx.x, c = threadIdx.x;
    int nbeg = blockIdx.y * 256;
    float s = scale[c], bi = bias[c], m = 0.0f;
    const float* p = H5 + ((size_t)b * 4096 + nbeg) * 256 + c;
    for (int n = 0; n < 256; ++n)
        m = fmaxf(m, fmaxf(p[(size_t)n * 256] * s + bi, 0.0f));
    atomicMax((unsigned int*)&z[b * 256 + c], __float_as_uint(m));
}

// ---------------------------------------------------------------------------
// pred[b,m,c] = o3[b, c*2048 + m]   (reshape(B,3,M).transpose(0,2,1))
// ---------------------------------------------------------------------------
__global__ void make_pred_kernel(const float* __restrict__ o3, float* __restrict__ pred) {
    int idx = blockIdx.x * 256 + threadIdx.x;
    if (idx >= 16 * 2048 * 3) return;
    int c = idx % 3;
    int m = (idx / 3) % 2048;
    int b = idx / (3 * 2048);
    pred[idx] = o3[(size_t)b * 6144 + c * 2048 + m];
}

// ---------------------------------------------------------------------------
// One-sided Chamfer: out[b,q] = min_r max(|Q_q|^2 + |R_r|^2 - 2 Q.R, 0)
// grid = (B, NQ/256), block 256; R tiled through LDS. NR multiple of 256.
// ---------------------------------------------------------------------------
__global__ void chamfer_min_kernel(const float* __restrict__ Q, const float* __restrict__ R,
                                   int NQ, int NR, float* __restrict__ out) {
    __shared__ float sx[256], sy[256], sz[256], s2[256];
    int b = blockIdx.x;
    int q = blockIdx.y * 256 + threadIdx.x;
    const float* qp = Q + ((size_t)b * NQ + q) * 3;
    float px = qp[0], py = qp[1], pz = qp[2];
    float p2 = px * px + py * py + pz * pz;
    float best = 3.4e38f;
    for (int t = 0; t < NR; t += 256) {
        __syncthreads();
        const float* rp = R + ((size_t)b * NR + t + threadIdx.x) * 3;
        float gx = rp[0], gy = rp[1], gz = rp[2];
        sx[threadIdx.x] = gx; sy[threadIdx.x] = gy; sz[threadIdx.x] = gz;
        s2[threadIdx.x] = gx * gx + gy * gy + gz * gz;
        __syncthreads();
        #pragma unroll 8
        for (int j = 0; j < 256; ++j) {
            float d = p2 + s2[j] - 2.0f * (px * sx[j] + py * sy[j] + pz * sz[j]);
            best = fminf(best, fmaxf(d, 0.0f));
        }
    }
    out[(size_t)b * NQ + q] = best;
}

// loss = mean_b( mean_m d1 + mean_n d2 ) ; equal counts per batch
__global__ void loss_kernel(const float* __restrict__ d1, const float* __restrict__ d2,
                            float* __restrict__ out) {
    __shared__ float red[256];
    float a = 0.0f;
    for (int i = threadIdx.x; i < 16 * 2048; i += 256) a += d1[i];
    a *= (1.0f / 2048.0f);
    float c = 0.0f;
    for (int i = threadIdx.x; i < 16 * 4096; i += 256) c += d2[i];
    c *= (1.0f / 4096.0f);
    red[threadIdx.x] = a + c;
    __syncthreads();
    for (int s = 128; s > 0; s >>= 1) {
        if (threadIdx.x < s) red[threadIdx.x] += red[threadIdx.x + s];
        __syncthreads();
    }
    if (threadIdx.x == 0) out[0] = red[0] * (1.0f / 16.0f);
}

// ---------------------------------------------------------------------------
// Host launcher
// ---------------------------------------------------------------------------
static void launch_gemm(const float* A, const float* ascale, const float* abias, int arelu,
                        int P, int Cin_real, int Cin_pad, const half_t* W16, int Cout,
                        const float* obias, int orelu, float* H,
                        float* cs, float* cq, int stats, hipStream_t stream) {
    int KP = Cin_pad < 128 ? Cin_pad : 128;     // all layer widths divide evenly
    dim3 grid((P + 63) / 64, Cout / 64);
    size_t sh = (size_t)128 * KP * sizeof(half_t) + 128 * sizeof(float);
    wmma_gemm_kernel<<<grid, 128, sh, stream>>>(A, ascale, abias, arelu, Cin_real, Cin_pad,
                                                KP, W16, Cout, P, obias, orelu, H, cs, cq, stats);
}

extern "C" void kernel_launch(void* const* d_in, const int* in_sizes, int n_in,
                              void* d_out, int out_size, void* d_ws, size_t ws_size,
                              hipStream_t stream) {
    const int P = 16 * 4096;                    // 65536 points
    const float* x  = (const float*)d_in[0];
    const float* gt = (const float*)d_in[1];
    const float* conv_w[5] = {(const float*)d_in[2],  (const float*)d_in[6],
                              (const float*)d_in[10], (const float*)d_in[14],
                              (const float*)d_in[18]};
    const float* conv_b[5] = {(const float*)d_in[3],  (const float*)d_in[7],
                              (const float*)d_in[11], (const float*)d_in[15],
                              (const float*)d_in[19]};
    const float* bn_g[5]   = {(const float*)d_in[4],  (const float*)d_in[8],
                              (const float*)d_in[12], (const float*)d_in[16],
                              (const float*)d_in[20]};
    const float* bn_b[5]   = {(const float*)d_in[5],  (const float*)d_in[9],
                              (const float*)d_in[13], (const float*)d_in[17],
                              (const float*)d_in[21]};
    const float* fc1_w = (const float*)d_in[22]; const float* fc1_b = (const float*)d_in[23];
    const float* fc2_w = (const float*)d_in[24]; const float* fc2_b = (const float*)d_in[25];
    const float* fc3_w = (const float*)d_in[26]; const float* fc3_b = (const float*)d_in[27];

    const int cinR[5] = {3, 64, 128, 128, 256};
    const int cinP[5] = {32, 64, 128, 128, 256};
    const int cout[5] = {64, 128, 128, 256, 256};

    // bump allocator over workspace
    char* ws = (char*)d_ws;
    size_t off = 0;
    auto alloc = [&](size_t bytes) -> char* {
        off = (off + 255) & ~(size_t)255;
        char* p = ws + off; off += bytes; return p;
    };

    half_t* w16[5];
    for (int i = 0; i < 5; ++i) w16[i] = (half_t*)alloc((size_t)cout[i] * cinP[i] * 2);
    half_t* wf1 = (half_t*)alloc((size_t)1024 * 256  * 2);
    half_t* wf2 = (half_t*)alloc((size_t)2048 * 1024 * 2);
    half_t* wf3 = (half_t*)alloc((size_t)6144 * 2048 * 2);

    float* stats = (float*)alloc(1664 * 4);     // sum+sumsq for all 5 layers, contiguous
    float* ssum[5], *ssq[5];
    {   float* p = stats;
        for (int i = 0; i < 5; ++i) { ssum[i] = p; p += cout[i]; ssq[i] = p; p += cout[i]; } }
    float* scl = (float*)alloc(1664 * 4);       // folded scale+bias per layer
    float* bscale[5], *bbias[5];
    {   float* p = scl;
        for (int i = 0; i < 5; ++i) { bscale[i] = p; p += cout[i]; bbias[i] = p; p += cout[i]; } }

    float* bufA = (float*)alloc((size_t)P * 256 * 4);   // 64 MB ping
    float* bufB = (float*)alloc((size_t)P * 256 * 4);   // 64 MB pong
    float* z    = (float*)alloc((size_t)16 * 256  * 4);
    float* o1   = (float*)alloc((size_t)16 * 1024 * 4);
    float* o2   = (float*)alloc((size_t)16 * 2048 * 4);
    float* o3   = (float*)alloc((size_t)16 * 6144 * 4);
    float* d1v  = (float*)alloc((size_t)16 * 2048 * 4);
    float* d2v  = (float*)alloc((size_t)16 * 4096 * 4);

    // --- weight conversion + zeroing ------------------------------------------
    auto cvt = [&](const float* src, half_t* dst, int Co, int CiR, int CiP) {
        int n = Co * CiP;
        convert_w_kernel<<<(n + 255) / 256, 256, 0, stream>>>(src, dst, Co, CiR, CiP);
    };
    for (int i = 0; i < 5; ++i) cvt(conv_w[i], w16[i], cout[i], cinR[i], cinP[i]);
    cvt(fc1_w, wf1, 1024, 256, 256);
    cvt(fc2_w, wf2, 2048, 1024, 1024);
    cvt(fc3_w, wf3, 6144, 2048, 2048);
    zero_f32_kernel<<<(1664 + 255) / 256, 256, 0, stream>>>(stats, 1664);
    zero_f32_kernel<<<(16 * 256 + 255) / 256, 256, 0, stream>>>(z, 16 * 256);

    // --- encoder: GEMM + stats, then BN fold, input transform fused forward ---
    const float invP = 1.0f / (float)P;
    float* src = nullptr;   // prev raw activations
    float* dst = bufA;
    for (int i = 0; i < 5; ++i) {
        const float* Ain = (i == 0) ? x : src;
        const float* asc = (i == 0) ? nullptr : bscale[i - 1];
        const float* abi = (i == 0) ? nullptr : bbias[i - 1];
        launch_gemm(Ain, asc, abi, /*arelu=*/(i != 0),
                    P, cinR[i], cinP[i], w16[i], cout[i],
                    conv_b[i], /*orelu=*/0, dst, ssum[i], ssq[i], /*stats=*/1, stream);
        bn_finalize_kernel<<<(cout[i] + 255) / 256, 256, 0, stream>>>(
            ssum[i], ssq[i], bn_g[i], bn_b[i], bscale[i], bbias[i], cout[i], invP);
        src = dst;
        dst = (dst == bufA) ? bufB : bufA;
    }
    // src now = raw h5 [P,256]; apply BN5+relu during max-pool
    maxpool_kernel<<<dim3(16, 16), 256, 0, stream>>>(src, bscale[4], bbias[4], z);

    // --- FC decoder (rows padded to one 64-row block tile) --------------------
    launch_gemm(z,  nullptr, nullptr, 0, 16, 256,  256,  wf1, 1024, fc1_b, 1, o1, nullptr, nullptr, 0, stream);
    launch_gemm(o1, nullptr, nullptr, 0, 16, 1024, 1024, wf2, 2048, fc2_b, 1, o2, nullptr, nullptr, 0, stream);
    launch_gemm(o2, nullptr, nullptr, 0, 16, 2048, 2048, wf3, 6144, fc3_b, 0, o3, nullptr, nullptr, 0, stream);

    // --- outputs: pred then Chamfer loss --------------------------------------
    float* pred = (float*)d_out;
    make_pred_kernel<<<(16 * 2048 * 3 + 255) / 256, 256, 0, stream>>>(o3, pred);
    chamfer_min_kernel<<<dim3(16, 2048 / 256), 256, 0, stream>>>(pred, gt, 2048, 4096, d1v);
    chamfer_min_kernel<<<dim3(16, 4096 / 256), 256, 0, stream>>>(gt, pred, 4096, 2048, d2v);
    loss_kernel<<<1, 256, 0, stream>>>(d1v, d2v, pred + 16 * 2048 * 3);
}
